// Attention_22402549416250
// MI455X (gfx1250) — matrix operand order repaired and checked by reading, not compiled, not run on hardware
//
#include <hip/hip_runtime.h>
#include <hip/hip_bf16.h>
#include <cstdint>
#include <cstddef>

// ---------------------------------------------------------------------------
// CDNA5 (gfx1250) wave32 WMMA implementation of adapter + QKV + head-realloc
// attention + projection.  All GEMMs use v_wmma_f32_16x16x32_bf16.
// Operand fetches are contiguous 128-bit loads; multiple independent
// accumulator chains per wave hide load latency behind the matrix pipe.
// ---------------------------------------------------------------------------

typedef __bf16 bf16_t;
typedef __attribute__((ext_vector_type(16))) __bf16 v16bf;
typedef __attribute__((ext_vector_type(8)))  __bf16 v8bf;
typedef __attribute__((ext_vector_type(8)))  float  v8f;

#define BT_    64
#define NTOK_  197
#define NPAD_  224     // seq padded to 7*32 for WMMA K-loops
#define RB_    13      // 13*16 = 208 computed rows per bt (covers 197)
#define C_     768
#define HADP_  192
#define C3_    2304
#define NH_    12
#define HD_    64
#define SCALE_ 0.125f  // 64^-0.5

__device__ __forceinline__ v8f zero8() {
    v8f z;
#pragma unroll
    for (int i = 0; i < 8; ++i) z[i] = 0.f;
    return z;
}

// A-matrix fragment: 16x32 bf16, row-major source, leading dim ld (mult of 8).
__device__ __forceinline__ v16bf load_A_frag(const bf16_t* base, int ld) {
    const int lane = threadIdx.x & 31;
    const int m    = lane & 15;
    const int half = lane >> 4;
    const bf16_t* rp = base + m * ld + half * 16;
    const v8bf lo = *(const v8bf*)(rp);
    const v8bf hi = *(const v8bf*)(rp + 8);
    return __builtin_shufflevector(lo, hi, 0, 1, 2, 3, 4, 5, 6, 7,
                                   8, 9, 10, 11, 12, 13, 14, 15);
}

// B-matrix fragment from B^T (N x K row-major, ld mult of 16): one 32B load.
__device__ __forceinline__ v16bf load_Bt_frag(const bf16_t* bt, int ldK) {
    const int lane = threadIdx.x & 31;
    const int n  = lane & 15;
    const int kh = (lane >> 4) << 4;
    return *(const v16bf*)(bt + (size_t)n * ldK + kh);
}

__device__ __forceinline__ v8f wmma_bf16(v16bf a, v16bf b, v8f c) {
    return __builtin_amdgcn_wmma_f32_16x16x32_bf16(false, a, false, b,
                                                   (short)0, c, false, false);
}

// ---------------------------------------------------------------------------
// Kernel 0: fp32 -> bf16 conversion WITH transpose (LDS-tiled, coalesced).
// s: [K][N] fp32 row-major  ->  d: [N][K] bf16 row-major.  K,N mult of 32.
// ---------------------------------------------------------------------------
__global__ void __launch_bounds__(256)
cvt_t_f32_bf16(const float* __restrict__ s, bf16_t* __restrict__ d, int K, int N) {
    __shared__ bf16_t tile[32][33];
    const int tilesN = N >> 5;
    const int tk = (blockIdx.x / tilesN) << 5;
    const int tn = (blockIdx.x % tilesN) << 5;
    const int lx = threadIdx.x & 31;
    const int ly = threadIdx.x >> 5;     // 0..7
#pragma unroll
    for (int yy = ly; yy < 32; yy += 8)
        tile[yy][lx] = (bf16_t)s[(size_t)(tk + yy) * N + tn + lx];
    __syncthreads();
#pragma unroll
    for (int yy = ly; yy < 32; yy += 8)
        d[(size_t)(tn + yy) * K + tk + lx] = tile[lx][yy];
}

// ---------------------------------------------------------------------------
// Kernel 1: fused adapter + residual:  xa = x + ((x@W1+b1)@W2+b2)  (bf16 out)
// One 16-row token tile per block, 4 waves.  W1,W2 given transposed (NxK).
// ---------------------------------------------------------------------------
__global__ void __launch_bounds__(128)
adapter_kernel(const float* __restrict__ x,
               const bf16_t* __restrict__ wA1T, const float* __restrict__ b1,
               const bf16_t* __restrict__ wA2T, const float* __restrict__ b2,
               bf16_t* __restrict__ xa)
{
    __shared__ __align__(16) bf16_t xt[16 * C_];     // 24 KB
    __shared__ __align__(16) bf16_t ht[16 * HADP_];  //  6 KB

    const int bid  = blockIdx.x;
    const int bt   = bid / RB_;
    const int rb   = bid % RB_;
    const int row0 = rb * 16;

    for (int i = threadIdx.x; i < 16 * C_; i += 128) {
        const int m = i / C_, c = i % C_;
        const int tok = row0 + m;
        const float vv = (tok < NTOK_) ? x[((size_t)bt * NTOK_ + tok) * C_ + c] : 0.f;
        xt[i] = (bf16_t)vv;
    }
    __syncthreads();

    const int wave = threadIdx.x >> 5;
    const int lane = threadIdx.x & 31;
    const int ln = lane & 15, lh = lane >> 4;

    // GEMM1: [16 x 768] @ [768 x 192] + b1 -> ht; 3 independent acc chains.
    {
        v8f acc[3] = { zero8(), zero8(), zero8() };
        const bf16_t* wp = wA1T + (size_t)(wave * 3) * 16 * C_;
        for (int kk = 0; kk < C_ / 32; ++kk) {
            const v16bf A = load_A_frag(xt + kk * 32, C_);
#pragma unroll
            for (int t = 0; t < 3; ++t)
                acc[t] = wmma_bf16(A, load_Bt_frag(wp + (size_t)t * 16 * C_ + kk * 32, C_), acc[t]);
        }
#pragma unroll
        for (int t = 0; t < 3; ++t) {
            const int col = (wave * 3 + t) * 16 + ln;
            const float bias = b1[col];
#pragma unroll
            for (int r = 0; r < 8; ++r)
                ht[(r + 8 * lh) * HADP_ + col] = (bf16_t)(acc[t][r] + bias);
        }
    }
    __syncthreads();

    // GEMM2: [16 x 192] @ [192 x 768] + b2 + residual; 3 groups x 4 accs.
    for (int g = 0; g < 3; ++g) {
        v8f acc[4] = { zero8(), zero8(), zero8(), zero8() };
        const int nt0 = wave * 12 + g * 4;
        const bf16_t* wp = wA2T + (size_t)nt0 * 16 * HADP_;
#pragma unroll
        for (int kk = 0; kk < HADP_ / 32; ++kk) {
            const v16bf A = load_A_frag(ht + kk * 32, HADP_);
#pragma unroll
            for (int t = 0; t < 4; ++t)
                acc[t] = wmma_bf16(A, load_Bt_frag(wp + (size_t)t * 16 * HADP_ + kk * 32, HADP_), acc[t]);
        }
#pragma unroll
        for (int t = 0; t < 4; ++t) {
            const int col = (nt0 + t) * 16 + ln;
            const float bias = b2[col];
#pragma unroll
            for (int r = 0; r < 8; ++r) {
                const int m = r + 8 * lh;
                const int tok = row0 + m;
                const float xv = (tok < NTOK_) ? x[((size_t)bt * NTOK_ + tok) * C_ + col] : 0.f;
                xa[((size_t)bt * NPAD_ + row0 + m) * C_ + col] = (bf16_t)(xv + acc[t][r] + bias);
            }
        }
    }
}

// ---------------------------------------------------------------------------
// Kernel 2: QKV GEMM  xa[16x768] @ wQKV[768x2304] + bias,
//   stores: q[bt][h][n][d], k[bt][h][n][d] (natural), vT[bt][h][d][n].
// 4 waves, 4 output tiles per wave (16 rows x 256 cols per block).
// ---------------------------------------------------------------------------
__global__ void __launch_bounds__(128)
qkv_kernel(const bf16_t* __restrict__ xa,
           const bf16_t* __restrict__ wQKVT, const float* __restrict__ qkvb,
           bf16_t* __restrict__ q, bf16_t* __restrict__ k, bf16_t* __restrict__ vT)
{
    __shared__ __align__(16) bf16_t at[16 * C_];     // 24 KB

    const int bid  = blockIdx.x;
    const int cg   = bid % 9;
    const int rblk = bid / 9;
    const int bt   = rblk / RB_;
    const int rb   = rblk % RB_;
    const int row0 = rb * 16;

    {   // contiguous bulk copy of the A tile
        const uint4* src = (const uint4*)(xa + ((size_t)bt * NPAD_ + row0) * C_);
        uint4* dst = (uint4*)at;
        for (int i = threadIdx.x; i < (16 * C_) / 8; i += 128) dst[i] = src[i];
    }
    __syncthreads();

    const int wave = threadIdx.x >> 5;
    const int lane = threadIdx.x & 31;
    const int ln = lane & 15, lh = lane >> 4;

    v8f acc[4] = { zero8(), zero8(), zero8(), zero8() };
    const int ct0 = cg * 16 + wave * 4;
    const bf16_t* wp = wQKVT + (size_t)ct0 * 16 * C_;

    for (int kk = 0; kk < C_ / 32; ++kk) {
        const v16bf A = load_A_frag(at + kk * 32, C_);
#pragma unroll
        for (int t = 0; t < 4; ++t)
            acc[t] = wmma_bf16(A, load_Bt_frag(wp + (size_t)t * 16 * C_ + kk * 32, C_), acc[t]);
    }

#pragma unroll
    for (int t = 0; t < 4; ++t) {
        const int ct   = ct0 + t;
        const int col0 = ct * 16;
        const int which = col0 / C_;
        const int rem   = col0 - which * C_;
        const int h  = rem / HD_;
        const int d0 = rem % HD_;
        const float bias = qkvb[col0 + ln];
        if (which == 0) {
            bf16_t* dst = q + (((size_t)bt * NH_ + h) * NPAD_ + row0) * HD_ + d0 + ln;
#pragma unroll
            for (int r = 0; r < 8; ++r)
                dst[(size_t)(r + 8 * lh) * HD_] = (bf16_t)(acc[t][r] + bias);
        } else if (which == 1) {   // K kept natural: [token][d] == B^T for S GEMM
            bf16_t* dst = k + (((size_t)bt * NH_ + h) * NPAD_ + row0) * HD_ + d0 + ln;
#pragma unroll
            for (int r = 0; r < 8; ++r)
                dst[(size_t)(r + 8 * lh) * HD_] = (bf16_t)(acc[t][r] + bias);
        } else {                   // V stored transposed: [d][token] == B^T for P@V
            bf16_t* dst = vT + (((size_t)bt * NH_ + h) * HD_ + d0 + ln) * NPAD_ + row0;
#pragma unroll
            for (int r = 0; r < 8; ++r)
                dst[r + 8 * lh] = (bf16_t)(acc[t][r] + bias);
        }
    }
}

// ---------------------------------------------------------------------------
// Kernel 3: attention with head reallocation as a GATHER on the source frame.
// One (bt, head, 16-row q tile) per block, 4 waves.
// S = Q*K^T (WMMA) -> masked softmax in LDS -> O = P*V (WMMA).
// ---------------------------------------------------------------------------
__global__ void __launch_bounds__(128)
attn_kernel(const bf16_t* __restrict__ q, const bf16_t* __restrict__ k,
            const bf16_t* __restrict__ vT, bf16_t* __restrict__ o)
{
    __shared__ __align__(16) float  sS[16 * NPAD_];  // 14 KB
    __shared__ __align__(16) bf16_t sP[16 * NPAD_];  //  7 KB
    __shared__ __align__(16) bf16_t sQ[16 * HD_];    //  2 KB

    const int bid = blockIdx.x;
    const int qt  = bid % RB_;
    const int hh  = (bid / RB_) % NH_;
    const int bt  = bid / (RB_ * NH_);
    const int b = bt / 8, f = bt % 8;
    int sf = f;                                    // head_reallocation gather
    if (hh < 2)      sf = (f > 0) ? f - 1 : 0;     // k'[f] = k[f-1]
    else if (hh < 4) sf = (f < 7) ? f + 1 : 7;     // k'[f] = k[f+1]
    const int sbt = b * 8 + sf;

    const bf16_t* qbase = q  + (((size_t)bt  * NH_ + hh) * NPAD_ + qt * 16) * HD_;
    const bf16_t* kbase = k  + ((size_t)sbt * NH_ + hh) * (size_t)NPAD_ * HD_;
    const bf16_t* vbase = vT + ((size_t)sbt * NH_ + hh) * (size_t)HD_ * NPAD_;

    for (int i = threadIdx.x; i < (16 * HD_) / 8; i += 128)
        ((uint4*)sQ)[i] = ((const uint4*)qbase)[i];
    __syncthreads();

    const int wave = threadIdx.x >> 5;
    const int lane = threadIdx.x & 31;
    const int ln = lane & 15, lh = lane >> 4;

    // ---- S = (Q @ K^T) * scale; Q frags (K=64 -> just 2) hoisted.
    {
        const v16bf qA0 = load_A_frag(sQ + 0,  HD_);
        const v16bf qA1 = load_A_frag(sQ + 32, HD_);
        for (int ct = wave; ct < NPAD_ / 16; ct += 4) {
            v8f acc = zero8();
            acc = wmma_bf16(qA0, load_Bt_frag(kbase + (size_t)(ct * 16) * HD_ + 0,  HD_), acc);
            acc = wmma_bf16(qA1, load_Bt_frag(kbase + (size_t)(ct * 16) * HD_ + 32, HD_), acc);
            const int col = ct * 16 + ln;
            const bool oob = (col >= NTOK_);
#pragma unroll
            for (int r = 0; r < 8; ++r)
                sS[(r + 8 * lh) * NPAD_ + col] = oob ? -3.0e38f : acc[r] * SCALE_;
        }
    }
    __syncthreads();

    // ---- softmax: 8 lanes per row, reduce with shfl_xor inside the wave
    {
        const int row = threadIdx.x >> 3;   // 0..15
        const int j0  = threadIdx.x & 7;
        float* srow = sS + row * NPAD_;
        float mx = -3.0e38f;
        for (int j = j0; j < NPAD_; j += 8) mx = fmaxf(mx, srow[j]);
        mx = fmaxf(mx, __shfl_xor(mx, 1, 32));
        mx = fmaxf(mx, __shfl_xor(mx, 2, 32));
        mx = fmaxf(mx, __shfl_xor(mx, 4, 32));
        float sum = 0.f;
        for (int j = j0; j < NPAD_; j += 8) {
            const float e = __expf(srow[j] - mx);
            srow[j] = e;
            sum += e;
        }
        sum += __shfl_xor(sum, 1, 32);
        sum += __shfl_xor(sum, 2, 32);
        sum += __shfl_xor(sum, 4, 32);
        const float inv = 1.f / sum;
        bf16_t* prow = sP + row * NPAD_;
        for (int j = j0; j < NPAD_; j += 8) prow[j] = (bf16_t)(srow[j] * inv);
    }
    __syncthreads();

    // ---- O = P @ V; B^T is vT's [d][token] layout.  One d tile per wave.
    {
        const int ct = wave;                 // 0..3 -> d columns
        v8f acc = zero8();
#pragma unroll
        for (int kk = 0; kk < NPAD_ / 32; ++kk) {
            const v16bf A = load_A_frag(sP + kk * 32, NPAD_);
            const v16bf B = load_Bt_frag(vbase + (size_t)(ct * 16) * NPAD_ + kk * 32, NPAD_);
            acc = wmma_bf16(A, B, acc);
        }
        bf16_t* obase = o + (((size_t)bt * NPAD_) + qt * 16) * C_ + hh * HD_ + ct * 16 + ln;
#pragma unroll
        for (int r = 0; r < 8; ++r)
            obase[(size_t)(r + 8 * lh) * C_] = (bf16_t)acc[r];
    }
}

// ---------------------------------------------------------------------------
// Kernel 4: projection  o[16x768] @ wP[768x768] + bias -> fp32 out (197 rows)
// 4 waves, 4 output tiles per wave (16 rows x 256 cols per block).
// ---------------------------------------------------------------------------
__global__ void __launch_bounds__(128)
proj_kernel(const bf16_t* __restrict__ o,
            const bf16_t* __restrict__ wPT, const float* __restrict__ pb,
            float* __restrict__ out)
{
    __shared__ __align__(16) bf16_t at[16 * C_];

    const int bid  = blockIdx.x;
    const int cg   = bid % 3;
    const int rblk = bid / 3;
    const int bt   = rblk / RB_;
    const int rb   = rblk % RB_;
    const int row0 = rb * 16;

    {
        const uint4* src = (const uint4*)(o + ((size_t)bt * NPAD_ + row0) * C_);
        uint4* dst = (uint4*)at;
        for (int i = threadIdx.x; i < (16 * C_) / 8; i += 128) dst[i] = src[i];
    }
    __syncthreads();

    const int wave = threadIdx.x >> 5;
    const int lane = threadIdx.x & 31;
    const int ln = lane & 15, lh = lane >> 4;

    v8f acc[4] = { zero8(), zero8(), zero8(), zero8() };
    const int ct0 = cg * 16 + wave * 4;
    const bf16_t* wp = wPT + (size_t)ct0 * 16 * C_;

    for (int kk = 0; kk < C_ / 32; ++kk) {
        const v16bf A = load_A_frag(at + kk * 32, C_);
#pragma unroll
        for (int t = 0; t < 4; ++t)
            acc[t] = wmma_bf16(A, load_Bt_frag(wp + (size_t)t * 16 * C_ + kk * 32, C_), acc[t]);
    }

#pragma unroll
    for (int t = 0; t < 4; ++t) {
        const int col = (ct0 + t) * 16 + ln;
        const float bias = pb[col];
#pragma unroll
        for (int r = 0; r < 8; ++r) {
            const int tok = row0 + r + 8 * lh;
            if (tok < NTOK_)
                out[((size_t)bt * NTOK_ + tok) * C_ + col] = acc[t][r] + bias;
        }
    }
}

// ---------------------------------------------------------------------------
// Workspace layout (bytes)
// ---------------------------------------------------------------------------
static constexpr size_t OFF_WA1  = 0;
static constexpr size_t OFF_WA2  = OFF_WA1  + (size_t)C_ * HADP_ * 2;
static constexpr size_t OFF_WQKV = OFF_WA2  + (size_t)HADP_ * C_ * 2;
static constexpr size_t OFF_WP   = OFF_WQKV + (size_t)C_ * C3_ * 2;
static constexpr size_t OFF_XA   = OFF_WP   + (size_t)C_ * C_ * 2;
static constexpr size_t SZ_XA    = (size_t)BT_ * NPAD_ * C_ * 2;
static constexpr size_t SZ_HEADS = (size_t)BT_ * NH_ * NPAD_ * HD_ * 2;
static constexpr size_t OFF_Q    = OFF_XA + SZ_XA;
static constexpr size_t OFF_K    = OFF_Q  + SZ_HEADS;
static constexpr size_t OFF_V    = OFF_K  + SZ_HEADS;
static constexpr size_t OFF_O    = OFF_V  + SZ_HEADS;   // total ~115.4 MB

extern "C" void kernel_launch(void* const* d_in, const int* in_sizes, int n_in,
                              void* d_out, int out_size, void* d_ws, size_t ws_size,
                              hipStream_t stream)
{
    const float* x      = (const float*)d_in[0];
    const float* a_w1   = (const float*)d_in[1];
    const float* a_b1   = (const float*)d_in[2];
    const float* a_w2   = (const float*)d_in[3];
    const float* a_b2   = (const float*)d_in[4];
    const float* qkv_w  = (const float*)d_in[5];
    const float* qkv_b  = (const float*)d_in[6];
    const float* proj_w = (const float*)d_in[7];
    const float* proj_b = (const float*)d_in[8];
    float* out = (float*)d_out;

    char* ws = (char*)d_ws;
    bf16_t* wA1T  = (bf16_t*)(ws + OFF_WA1);
    bf16_t* wA2T  = (bf16_t*)(ws + OFF_WA2);
    bf16_t* wQKVT = (bf16_t*)(ws + OFF_WQKV);
    bf16_t* wPT   = (bf16_t*)(ws + OFF_WP);
    bf16_t* xa    = (bf16_t*)(ws + OFF_XA);
    bf16_t* qb    = (bf16_t*)(ws + OFF_Q);
    bf16_t* kb    = (bf16_t*)(ws + OFF_K);
    bf16_t* vTb   = (bf16_t*)(ws + OFF_V);
    bf16_t* ob    = (bf16_t*)(ws + OFF_O);

    // 0) weights -> bf16, transposed to NxK (hot in 192MB L2 afterwards)
    cvt_t_f32_bf16<<<(C_ / 32) * (HADP_ / 32), 256, 0, stream>>>(a_w1,   wA1T,  C_,    HADP_);
    cvt_t_f32_bf16<<<(HADP_ / 32) * (C_ / 32), 256, 0, stream>>>(a_w2,   wA2T,  HADP_, C_);
    cvt_t_f32_bf16<<<(C_ / 32) * (C3_ / 32),   256, 0, stream>>>(qkv_w,  wQKVT, C_,    C3_);
    cvt_t_f32_bf16<<<(C_ / 32) * (C_ / 32),    256, 0, stream>>>(proj_w, wPT,   C_,    C_);

    // 1) adapter + residual
    adapter_kernel<<<BT_ * RB_, 128, 0, stream>>>(x, wA1T, a_b1, wA2T, a_b2, xa);

    // 2) QKV projection (q / k natural / v transposed, padded layouts)
    qkv_kernel<<<BT_ * RB_ * 9, 128, 0, stream>>>(xa, wQKVT, qkv_b, qb, kb, vTb);

    // 3) attention with head-reallocation gather
    attn_kernel<<<BT_ * NH_ * RB_, 128, 0, stream>>>(qb, kb, vTb, ob);

    // 4) output projection -> fp32
    proj_kernel<<<BT_ * RB_ * 3, 128, 0, stream>>>(ob, wPT, proj_b, out);
}
